// SelfAttention_38800734552105
// MI455X (gfx1250) — matrix operand order repaired and checked by reading, not compiled, not run on hardware
//
#include <hip/hip_runtime.h>

typedef __attribute__((ext_vector_type(16))) _Float16 v16h;
typedef __attribute__((ext_vector_type(8)))  _Float16 v8h;
typedef __attribute__((ext_vector_type(8)))  float    v8f;
typedef __attribute__((ext_vector_type(4)))  unsigned int u32x4;
typedef __attribute__((ext_vector_type(4)))  int      i32x4;

#define HID   1024
#define BATCH 4
#define SEQ   2048
#define MTOT  (BATCH*SEQ)      // 8192

#define BM 128
#define BN 256
#define BK 32
#define APAD 8
#define BPAD 8
#define NBUF 3                 // 3-deep async pipeline

// ---- CDNA5 async global->LDS copies (guarded; falls back to sync staging) ----
#if defined(__gfx1250__) && __has_builtin(__builtin_amdgcn_global_load_async_to_lds_b128) && __has_builtin(__builtin_amdgcn_s_wait_asynccnt)
#define HAVE_ASYNC 1
#else
#define HAVE_ASYNC 0
#endif

#if HAVE_ASYNC
__device__ __forceinline__ void cp_async16(const _Float16* g, _Float16* l) {
  __builtin_amdgcn_global_load_async_to_lds_b128(
      (__attribute__((address_space(1))) i32x4*)(g),
      (__attribute__((address_space(3))) i32x4*)(l),
      /*offset*/ 0, /*cpol*/ 0);
}
#endif

// ---------------------------------------------------------------- converts
__global__ __launch_bounds__(256) void cvt_f32_f16(const float* __restrict__ in,
                                                   _Float16* __restrict__ out, int n) {
  int i = blockIdx.x * blockDim.x + threadIdx.x;
  int stride = gridDim.x * blockDim.x;
  for (; i < n; i += stride) out[i] = (_Float16)in[i];
}

// Wt[d][e] = W[e][d], f32 -> f16, LDS-tiled transpose
__global__ __launch_bounds__(256) void transpose_cvt(const float* __restrict__ W,
                                                     _Float16* __restrict__ Wt) {
  __shared__ float tile[32][33];
  int tx = threadIdx.x, ty = threadIdx.y;           // blockDim = (32,8)
  int e0 = blockIdx.y * 32, d0 = blockIdx.x * 32;
  #pragma unroll
  for (int j = 0; j < 32; j += 8)
    tile[ty + j][tx] = W[(long)(e0 + ty + j) * HID + d0 + tx];
  __syncthreads();
  #pragma unroll
  for (int j = 0; j < 32; j += 8)
    Wt[(long)(d0 + ty + j) * HID + e0 + tx] = (_Float16)tile[tx][ty + j];
}

// ---------------------------------------------------------------- WMMA GEMM
// C = A(f16,row-major,MxK) * B(f16,row-major,KxN) [+ bias]
// Block tile 128x256, 8 waves, each wave 64x64 = 16 wmma tiles.
// Triple-buffered LDS; 2-deep async global->LDS prefetch when available.
// EPI 0: f16 out, +bias, row-major     (Q,V projections)
// EPI 1: f16 out, +bias, transposed: C[n*ldc + m]   (K stored as Kt[d][row])
// EPI 2: f32 out, no bias, row-major   (scores, attn*V)
template <int EPI>
__global__ __launch_bounds__(256) void gemm_wmma(
    const _Float16* __restrict__ A, const _Float16* __restrict__ B,
    void* __restrict__ Cv, const float* __restrict__ bias,
    int M, int N, int K, int lda, int ldb, int ldc,
    long batchA, long batchB, long batchC)
{
  __shared__ _Float16 As[NBUF][BM][BK + APAD];   // 3 x 10.0 KB
  __shared__ _Float16 Bs[NBUF][BK][BN + BPAD];   // 3 x 16.5 KB

  const int z = blockIdx.z;
  A += (long)z * batchA;
  B += (long)z * batchB;

  const int tid  = threadIdx.x;
  const int lane = tid & 31;
  const int wave = tid >> 5;          // 8 waves
  const int waveM = wave >> 2;        // 0..1  -> 64 rows each
  const int waveN = wave & 3;         // 0..3  -> 64 cols each
  const int m0 = blockIdx.y * BM;
  const int n0 = blockIdx.x * BN;
  const int lane16   = lane & 15;
  const int laneHalf = lane >> 4;

  v8f acc[4][4];
  #pragma unroll
  for (int m = 0; m < 4; ++m)
    #pragma unroll
    for (int n = 0; n < 4; ++n) acc[m][n] = (v8f){};

  union Frag { v16h v; v8h h[2]; };

  // ---- per-thread transfer slots, global pointers advanced incrementally ----
  // A: 128x32 halves = 512 x 16B chunks, 2 per thread
  int aRow[2], aCg[2];
  const _Float16* aP[2];
  #pragma unroll
  for (int i = 0; i < 2; ++i) {
    int idx = tid + i * 256;
    aRow[i] = idx >> 2;
    aCg[i]  = (idx & 3) * 8;
    aP[i]   = A + (long)(m0 + aRow[i]) * lda + aCg[i];
  }
  // B: 32x256 halves = 1024 x 16B chunks, 4 per thread
  int bRow[4], bCg[4];
  const _Float16* bP[4];
  #pragma unroll
  for (int i = 0; i < 4; ++i) {
    int idx = tid + i * 256;
    bRow[i] = idx >> 5;
    bCg[i]  = (idx & 31) * 8;
    bP[i]   = B + (long)bRow[i] * ldb + n0 + bCg[i];
  }
  const long bStep = (long)BK * ldb;

  // issue the next sequential K-tile into LDS buffer `buf`, advance pointers
  auto issue_tile = [&](int buf) {
#if HAVE_ASYNC
    #pragma unroll
    for (int i = 0; i < 2; ++i) {
      cp_async16(aP[i], &As[buf][aRow[i]][aCg[i]]);
      aP[i] += BK;
    }
    #pragma unroll
    for (int i = 0; i < 4; ++i) {
      cp_async16(bP[i], &Bs[buf][bRow[i]][bCg[i]]);
      bP[i] += bStep;
    }
#else
    u32x4 ra[2], rb[4];
    #pragma unroll
    for (int i = 0; i < 2; ++i) { ra[i] = *(const u32x4*)aP[i]; aP[i] += BK; }
    #pragma unroll
    for (int i = 0; i < 4; ++i) { rb[i] = *(const u32x4*)bP[i]; bP[i] += bStep; }
    #pragma unroll
    for (int i = 0; i < 2; ++i) *(u32x4*)&As[buf][aRow[i]][aCg[i]] = ra[i];
    #pragma unroll
    for (int i = 0; i < 4; ++i) *(u32x4*)&Bs[buf][bRow[i]][bCg[i]] = rb[i];
#endif
  };

  const int T = K / BK;
  issue_tile(0);
  if (T > 1) issue_tile(1);

  int cur = 0, nxt = 2;
  for (int t = 0; t < T; ++t) {
#if HAVE_ASYNC
    if (t + 1 < T) __builtin_amdgcn_s_wait_asynccnt(6);  // tile t landed; tile t+1 may fly
    else           __builtin_amdgcn_s_wait_asynccnt(0);
#endif
    __syncthreads();              // all waves' tile-t writes visible; buf `nxt` fully consumed
    if (t + 2 < T) {
      issue_tile(nxt);
      nxt = (nxt + 1 == NBUF) ? 0 : nxt + 1;
    }

    Frag af[4], bf[4];
    #pragma unroll
    for (int m = 0; m < 4; ++m) {
      int r  = waveM * 64 + m * 16 + lane16;
      int kg = laneHalf * 8;      // lanes<16: K0..7/16..23, lanes>=16: K8..15/24..31
      af[m].h[0] = *(const v8h*)&As[cur][r][kg];
      af[m].h[1] = *(const v8h*)&As[cur][r][kg + 16];
    }
    #pragma unroll
    for (int n = 0; n < 4; ++n) {
      int c = waveN * 64 + n * 16 + lane16; // lane holds N=lane16, K in table order (same as A)
      #pragma unroll
      for (int s = 0; s < 8; ++s) {
        bf[n].v[s]     = Bs[cur][laneHalf * 8 + s][c];
        bf[n].v[s + 8] = Bs[cur][16 + laneHalf * 8 + s][c];
      }
    }
    #pragma unroll
    for (int m = 0; m < 4; ++m)
      #pragma unroll
      for (int n = 0; n < 4; ++n)
        acc[m][n] = __builtin_amdgcn_wmma_f32_16x16x32_f16(
            false, af[m].v, false, bf[n].v, (short)0, acc[m][n], false, false);

    cur = (cur + 1 == NBUF) ? 0 : cur + 1;
  }

  // epilogue: C frag VGPR r -> M=r+8*laneHalf, N=lane16
  #pragma unroll
  for (int m = 0; m < 4; ++m) {
    #pragma unroll
    for (int n = 0; n < 4; ++n) {
      int gcol = n0 + waveN * 64 + n * 16 + lane16;
      float bvv = (EPI == 2) ? 0.0f : bias[gcol];
      #pragma unroll
      for (int r = 0; r < 8; ++r) {
        int grow = m0 + waveM * 64 + m * 16 + r + laneHalf * 8;
        float v = acc[m][n][r] + bvv;
        if (EPI == 0) {
          ((_Float16*)Cv)[(long)z * batchC + (long)grow * ldc + gcol] = (_Float16)v;
        } else if (EPI == 1) {
          ((_Float16*)Cv)[(long)z * batchC + (long)gcol * ldc + grow] = (_Float16)v;
        } else {
          ((float*)Cv)[(long)z * batchC + (long)grow * ldc + gcol] = v;
        }
      }
    }
  }
}

// ---------------------------------------------------------------- softmax
// one 256-thread block per row of 2048 f32 scores -> f16 probabilities
__global__ __launch_bounds__(256) void softmax_rows(const float* __restrict__ S,
                                                    _Float16* __restrict__ P) {
  const long row = blockIdx.x;
  const float* in = S + row * SEQ;
  _Float16*   out = P + row * SEQ;
  const int tid = threadIdx.x;
  const int lane = tid & 31, wv = tid >> 5;

  float vals[8];
  float mx = -3.4e38f;
  #pragma unroll
  for (int i = 0; i < 8; ++i) { vals[i] = in[tid + i * 256]; mx = fmaxf(mx, vals[i]); }
  #pragma unroll
  for (int off = 16; off > 0; off >>= 1) mx = fmaxf(mx, __shfl_xor(mx, off, 32));

  __shared__ float redM[8];
  __shared__ float redS[8];
  if (lane == 0) redM[wv] = mx;
  __syncthreads();
  mx = redM[0];
  #pragma unroll
  for (int w = 1; w < 8; ++w) mx = fmaxf(mx, redM[w]);

  float sum = 0.0f;
  #pragma unroll
  for (int i = 0; i < 8; ++i) { vals[i] = __expf(vals[i] - mx); sum += vals[i]; }
  #pragma unroll
  for (int off = 16; off > 0; off >>= 1) sum += __shfl_xor(sum, off, 32);
  if (lane == 0) redS[wv] = sum;
  __syncthreads();
  sum = 0.0f;
  #pragma unroll
  for (int w = 0; w < 8; ++w) sum += redS[w];

  float inv = 1.0f / sum;
  #pragma unroll
  for (int i = 0; i < 8; ++i) out[tid + i * 256] = (_Float16)(vals[i] * inv);
}

// ---------------------------------------------------------------- launcher
extern "C" void kernel_launch(void* const* d_in, const int* in_sizes, int n_in,
                              void* d_out, int out_size, void* d_ws, size_t ws_size,
                              hipStream_t stream) {
  (void)in_sizes; (void)n_in; (void)out_size; (void)ws_size;
  const float* x  = (const float*)d_in[0];
  const float* Wq = (const float*)d_in[1];
  const float* bq = (const float*)d_in[2];
  const float* Wk = (const float*)d_in[3];
  const float* bk = (const float*)d_in[4];
  const float* Wv = (const float*)d_in[5];
  const float* bv = (const float*)d_in[6];
  float* out = (float*)d_out;

  // workspace layout (bytes), all MB-aligned; total 166 MB
  char* ws = (char*)d_ws;
  const size_t MB = 1024ull * 1024ull;
  _Float16* xh   = (_Float16*)(ws + 0);        // 16 MB  [8192][1024]
  _Float16* wtq  = (_Float16*)(ws + 16 * MB);  //  2 MB  [1024 d][1024 e]
  _Float16* wtk  = (_Float16*)(ws + 18 * MB);
  _Float16* wtv  = (_Float16*)(ws + 20 * MB);
  _Float16* qh   = (_Float16*)(ws + 22 * MB);  // 16 MB  [8192][1024]
  _Float16* kt   = (_Float16*)(ws + 38 * MB);  // 16 MB  [1024 d][8192 = b*2048+t]
  _Float16* vh   = (_Float16*)(ws + 54 * MB);  // 16 MB  [8192][1024]
  float*    sc   = (float*)   (ws + 70 * MB);  // 64 MB  [4][2048][2048] f32
  _Float16* attn = (_Float16*)(ws + 134 * MB); // 32 MB  [4][2048][2048] f16

  // 1) x -> f16
  cvt_f32_f16<<<4096, 256, 0, stream>>>(x, xh, MTOT * HID);

  // 2) W -> f16 transposed (Wt[d][e] = W[e][d])
  {
    dim3 tb(32, 8), tg(HID / 32, HID / 32);
    transpose_cvt<<<tg, tb, 0, stream>>>(Wq, wtq);
    transpose_cvt<<<tg, tb, 0, stream>>>(Wk, wtk);
    transpose_cvt<<<tg, tb, 0, stream>>>(Wv, wtv);
  }

  // 3) QKV projections: [8192,1024] = xh @ Wt + b
  {
    dim3 g(HID / BN, MTOT / BM, 1);
    gemm_wmma<0><<<g, 256, 0, stream>>>(xh, wtq, qh, bq,
        MTOT, HID, HID, HID, HID, HID, 0, 0, 0);
    // K stored transposed: Kt[d][b*2048+t], ldc = MTOT
    gemm_wmma<1><<<g, 256, 0, stream>>>(xh, wtk, kt, bk,
        MTOT, HID, HID, HID, HID, MTOT, 0, 0, 0);
    gemm_wmma<0><<<g, 256, 0, stream>>>(xh, wtv, vh, bv,
        MTOT, HID, HID, HID, HID, HID, 0, 0, 0);
  }

  // 4) scores[b] = Q[b] @ K[b]^T  -> f32 [2048,2048] per batch
  {
    dim3 g(SEQ / BN, SEQ / BM, BATCH);
    gemm_wmma<2><<<g, 256, 0, stream>>>(qh, kt, sc, nullptr,
        SEQ, SEQ, HID,
        /*lda*/ HID, /*ldb (Kt row stride)*/ MTOT, /*ldc*/ SEQ,
        /*batchA*/ (long)SEQ * HID, /*batchB (col offset into Kt)*/ (long)SEQ,
        /*batchC*/ (long)SEQ * SEQ);
  }

  // 5) softmax rows -> f16 attn
  softmax_rows<<<BATCH * SEQ, 256, 0, stream>>>(sc, attn);

  // 6) out[b] = attn[b] @ V[b]  -> f32 [2048,1024] per batch
  {
    dim3 g(HID / BN, SEQ / BM, BATCH);
    gemm_wmma<2><<<g, 256, 0, stream>>>(attn, vh, out, nullptr,
        SEQ, HID, SEQ,
        /*lda*/ SEQ, /*ldb*/ HID, /*ldc*/ HID,
        /*batchA*/ (long)SEQ * SEQ, /*batchB*/ (long)SEQ * HID,
        /*batchC*/ (long)SEQ * HID);
  }
}
